// SPEFastTransformerDecoder_57767310131322
// MI455X (gfx1250) — matrix operand order repaired and checked by reading, not compiled, not run on hardware
//
#include <hip/hip_runtime.h>
#include <hip/hip_bf16.h>

typedef __bf16 bf16_t;
typedef __attribute__((ext_vector_type(16))) __bf16 v16bf;
typedef __attribute__((ext_vector_type(8)))  __bf16 v8bf;
typedef __attribute__((ext_vector_type(4)))  __bf16 v4bf;
typedef __attribute__((ext_vector_type(8)))  float  v8f;

namespace cfg {
constexpr int B = 4, S = 1024, L = 4, H = 8, DM = 512, DF = 2048;
constexpr int DH = 64, NS = 5, R = 32, M = 128;
constexpr int NT = B * S;  // 4096 token rows
}

// ---------------------------------------------------------------------------
// helpers
// ---------------------------------------------------------------------------
__device__ inline v8f vzero8() {
  v8f z;
#pragma unroll
  for (int i = 0; i < 8; ++i) z[i] = 0.f;
  return z;
}

__device__ inline v8f wmma_bf16(v16bf a, v16bf b, v8f c) {
  // D = A(16x32 bf16) * B(32x16 bf16) + C(16x16 f32)
  return __builtin_amdgcn_wmma_f32_16x16x32_bf16(false, a, false, b, (short)0, c,
                                                 false, false);
}

// A-matrix fragment (16x32 bf16) from row-major LDS, row stride rs (elements,
// rs % 8 == 0). Lane layout per CDNA5 ISA 7.12.2: lane m=lane&15,
// a[0..7] = row[kh..kh+7], a[8..15] = row[16+kh..16+kh+7], kh = (lane>>4)*8.
// Both halves are 16B-aligned contiguous -> 2x ds_load_b128.
__device__ inline v16bf ld_a_rowmajor(const bf16_t* base, int rs) {
  const int lane = threadIdx.x & 31;
  const bf16_t* p = base + (lane & 15) * rs + ((lane >> 4) * 8);
  v8bf lo = *(const v8bf*)(p);
  v8bf hi = *(const v8bf*)(p + 16);
  return __builtin_shufflevector(lo, hi, 0, 1, 2, 3, 4, 5, 6, 7, 8, 9, 10, 11,
                                 12, 13, 14, 15);
}

// B-matrix fragment (32x16 bf16) from K-contiguous LDS: element (k,n) at
// base[n*ns + k], ns % 8 == 0. Lane n=lane&15 needs b[0..7] = k kh..kh+7,
// b[8..15] = k 16+kh..16+kh+7 (kh = (lane>>4)*8) -> 2x ds_load_b128.
__device__ inline v16bf ld_b_kcont(const bf16_t* base, int ns) {
  const int lane = threadIdx.x & 31;
  const bf16_t* p = base + (lane & 15) * ns + ((lane >> 4) * 8);
  v8bf lo = *(const v8bf*)(p);
  v8bf hi = *(const v8bf*)(p + 16);
  return __builtin_shufflevector(lo, hi, 0, 1, 2, 3, 4, 5, 6, 7, 8, 9, 10, 11,
                                 12, 13, 14, 15);
}

__device__ inline float sigmoidf_(float x) { return 1.f / (1.f + __expf(-x)); }

// ---------------------------------------------------------------------------
// cast f32 -> bf16
// ---------------------------------------------------------------------------
__global__ __launch_bounds__(256) void cast_f32_bf16_kernel(const float* __restrict__ in,
                                                            bf16_t* __restrict__ out,
                                                            int n) {
  int i = blockIdx.x * 256 + threadIdx.x;
  if (i < n) out[i] = (bf16_t)in[i];
}

// ---------------------------------------------------------------------------
// SineSPE positional code: qbar/kbar (S,H,DH,R) for one layer.
// ---------------------------------------------------------------------------
__global__ __launch_bounds__(256) void spe_code_kernel(const float* __restrict__ freqs,
                                                       const float* __restrict__ offs,
                                                       const float* __restrict__ gains,
                                                       const float* __restrict__ noise,
                                                       float* __restrict__ qbar,
                                                       float* __restrict__ kbar) {
  using namespace cfg;
  const int hd = blockIdx.x;  // h*DH + d
  const int t = threadIdx.x;
  __shared__ float fz[NS], oz[NS];
  __shared__ float zz[2 * NS][R];

  if (t < NS) {
    fz[t] = sigmoidf_(freqs[hd * NS + t]) * 0.5f;
    oz[t] = offs[hd * NS + t];
  }
  const float inv_sqrt_2ns = 0.31622776601f;  // 1/sqrt(10)
  for (int i = t; i < 2 * NS * R; i += 256) {
    int kk = i / R;
    float gsp = log1pf(__expf(gains[hd * NS + (kk >> 1)]));  // softplus, upsampled
    zz[kk][i % R] = noise[(size_t)hd * (2 * NS * R) + i] * gsp * inv_sqrt_2ns;
  }
  __syncthreads();

  const float inv_scale = 0.14865088937f;  // (R*DH)^-0.25 = 2048^-0.25
  for (int s = t; s < S; s += 256) {
    float omq[2 * NS], omk[2 * NS];
#pragma unroll
    for (int i = 0; i < NS; ++i) {
      float ph = 6.28318530718f * fz[i] * (float)s;
      float sq, cq;
      __sincosf(ph + oz[i], &sq, &cq);
      omq[2 * i] = cq; omq[2 * i + 1] = sq;
      float sk_, ck_;
      __sincosf(ph, &sk_, &ck_);
      omk[2 * i] = ck_; omk[2 * i + 1] = sk_;
    }
    float* qo = qbar + ((size_t)s * (H * DH) + hd) * R;
    float* ko = kbar + ((size_t)s * (H * DH) + hd) * R;
#pragma unroll 4
    for (int r = 0; r < R; ++r) {
      float aq = 0.f, ak = 0.f;
#pragma unroll
      for (int kk = 0; kk < 2 * NS; ++kk) {
        aq += omq[kk] * zz[kk][r];
        ak += omk[kk] * zz[kk][r];
      }
      qo[r] = aq * inv_scale;
      ko[r] = ak * inv_scale;
    }
  }
}

// ---------------------------------------------------------------------------
// bf16 WMMA GEMM: C(Mrows x N) = A(Mrows x K) * W(K x N) + bias, opt ReLU,
// output f32 or bf16. 128x128 tile per block; B staged transposed so
// fragments are K-contiguous (ds_load_b128).
// ---------------------------------------------------------------------------
template <int RELU, int OUTBF16>
__global__ __launch_bounds__(256) void gemm_kernel(const bf16_t* __restrict__ A,
                                                   const bf16_t* __restrict__ W,
                                                   const float* __restrict__ bias,
                                                   float* __restrict__ Cf,
                                                   bf16_t* __restrict__ Cb,
                                                   int Mrows, int N, int K) {
  constexpr int LDA = 40;   // 32 + pad (elements; 80B rows, 16B aligned)
  constexpr int LDBT = 40;  // k-stride of transposed B rows
  __shared__ bf16_t As[128 * LDA];
  __shared__ bf16_t BsT[128 * LDBT];  // BsT[n][k]

  const int t = threadIdx.x;
  const int wave = t >> 5, lane = t & 31;
  const int rowBase = blockIdx.x * 128;
  const int colBase = blockIdx.y * 128;
  const int wr = (wave >> 1) * 32;
  const int wc = (wave & 1) * 64;

  v8f acc[2][4];
#pragma unroll
  for (int i = 0; i < 2; ++i)
#pragma unroll
    for (int j = 0; j < 4; ++j) acc[i][j] = vzero8();

  for (int k0 = 0; k0 < K; k0 += 32) {
    // A: 128x32 in 16B chunks, straight copy
    for (int i = t; i < 512; i += 256) {
      int r = i >> 2, c8 = (i & 3) * 8;
      *(v8bf*)(As + r * LDA + c8) =
          *(const v8bf*)(A + (size_t)(rowBase + r) * K + k0 + c8);
    }
    // B: 32x128 read in 16B chunks, scattered transposed into BsT
    for (int i = t; i < 512; i += 256) {
      int r = i >> 4, c8 = (i & 15) * 8;
      v8bf w8 = *(const v8bf*)(W + (size_t)(k0 + r) * N + colBase + c8);
#pragma unroll
      for (int j = 0; j < 8; ++j) BsT[(c8 + j) * LDBT + r] = w8[j];
    }
    if (k0 + 32 < K) {  // pull next K-slab (global_prefetch_b8)
      __builtin_prefetch(&A[(size_t)(rowBase + (t >> 1)) * K + k0 + 32], 0, 1);
      __builtin_prefetch(&W[(size_t)(k0 + 32 + (t >> 3)) * N + colBase], 0, 1);
    }
    __syncthreads();

    v16bf a0 = ld_a_rowmajor(As + wr * LDA, LDA);
    v16bf a1 = ld_a_rowmajor(As + (wr + 16) * LDA, LDA);
#pragma unroll
    for (int ni = 0; ni < 4; ++ni) {
      v16bf bm = ld_b_kcont(BsT + (wc + ni * 16) * LDBT, LDBT);
      acc[0][ni] = wmma_bf16(a0, bm, acc[0][ni]);
      acc[1][ni] = wmma_bf16(a1, bm, acc[1][ni]);
    }
    __syncthreads();
  }

  const int n = lane & 15, mo = (lane >> 4) * 8;
#pragma unroll
  for (int mi = 0; mi < 2; ++mi)
#pragma unroll
    for (int ni = 0; ni < 4; ++ni) {
#pragma unroll
      for (int r = 0; r < 8; ++r) {
        int row = rowBase + wr + mi * 16 + mo + r;
        int col = colBase + wc + ni * 16 + n;
        float v = acc[mi][ni][r] + bias[col];
        if (RELU) v = fmaxf(v, 0.f);
        if (OUTBF16)
          Cb[(size_t)row * N + col] = (bf16_t)v;
        else
          Cf[(size_t)row * N + col] = v;
      }
    }
}

// ---------------------------------------------------------------------------
// Gated SPE filter: qh/kh (B,S,H,2R). One thread per (b,s,h,rr).
// ---------------------------------------------------------------------------
__global__ __launch_bounds__(256) void spe_filter_kernel(const float* __restrict__ qf,
                                                         const float* __restrict__ kf,
                                                         const float* __restrict__ qbar,
                                                         const float* __restrict__ kbar,
                                                         const float* __restrict__ gate,
                                                         const float* __restrict__ gnoise,
                                                         float* __restrict__ qh,
                                                         float* __restrict__ kh) {
  using namespace cfg;
  const size_t idx = (size_t)blockIdx.x * 256 + threadIdx.x;
  const int rr = (int)(idx & 63);
  const size_t bsh = idx >> 6;  // (b*S+s)*H + h
  const int h = (int)(bsh % H);
  const int s = (int)((bsh / H) % S);
  const float* q = qf + bsh * DH;
  const float* k = kf + bsh * DH;
  const float inv4 = 0.14865088937f;  // (DH*R)^-0.25
  float aq = 0.f, ak = 0.f;

  if (rr < R) {
    const float* qb = qbar + ((size_t)(s * H + h) * DH) * R + rr;
    const float* kb = kbar + ((size_t)(s * H + h) * DH) * R + rr;
#pragma unroll 8
    for (int d = 0; d < DH; ++d) {
      float g = sigmoidf_(gate[h * DH + d]);
      float w = sqrtf(1.f - g);
      aq += q[d] * (w * qb[(size_t)d * R]);
      ak += k[d] * (w * kb[(size_t)d * R]);
    }
  } else {
    const int r = rr - R;
#pragma unroll 8
    for (int d = 0; d < DH; ++d) {
      float g = sigmoidf_(gate[h * DH + d]);
      float bia = sqrtf(g) * gnoise[(size_t)(h * DH + d) * R + r] * inv4;
      aq += q[d] * bia;
      ak += k[d] * bia;
    }
  }
  qh[bsh * 64 + rr] = aq;
  kh[bsh * 64 + rr] = ak;
}

// ---------------------------------------------------------------------------
// FAVOR+ feature map: Phi (B,S,H,M). One thread per (b,s,h,j).
// ---------------------------------------------------------------------------
__global__ __launch_bounds__(256) void favor_kernel(const float* __restrict__ Xh,
                                                    const float* __restrict__ omega,
                                                    float* __restrict__ Phi) {
  using namespace cfg;
  const size_t idx = (size_t)blockIdx.x * 256 + threadIdx.x;
  const int j = (int)(idx & 63);
  const size_t bsh = idx >> 6;
  const float* x = Xh + bsh * 64;
  const float sc = 0.35355339059f;  // 64^-0.25
  float u = 0.f, s2 = 0.f;
#pragma unroll 8
  for (int e = 0; e < 64; ++e) {
    float xe = x[e] * sc;
    u += xe * omega[e * 64 + j];
    s2 += xe * xe;
  }
  const float hn = 0.5f * s2;
  const float inv = 0.08838834765f;  // 1/sqrt(M)
  float* ph = Phi + bsh * M;
  ph[j]      = __expf(u - hn) * inv;
  ph[64 + j] = __expf(-u - hn) * inv;
}

// ---------------------------------------------------------------------------
// Chunked causal linear attention (chunk C=32), WMMA throughout.
// Block per (b,h); 8 waves. Skv (128x64 f32) lives in WMMA accumulators
// (wave w owns feature rows 16w..16w+15). All LDS operands laid out so
// fragments are contiguous (ds_load_b128).
// ---------------------------------------------------------------------------
__global__ __launch_bounds__(256) void attn_kernel(const float* __restrict__ Qf,
                                                   const float* __restrict__ Kf,
                                                   const float* __restrict__ Vf,
                                                   float* __restrict__ Out) {
  using namespace cfg;
  constexpr int KP = 40;   // kcT row stride
  constexpr int VP = 40;   // vcT row stride
  constexpr int MP = 136;  // skvbT row stride (128 + pad)
  const int bh = blockIdx.x;
  const int b = bh / H, h = bh % H;
  const int t = threadIdx.x;
  const int wave = t >> 5, lane = t & 31;

  __shared__ bf16_t qc[32 * M];       // row-major 32x128 (A operand)
  __shared__ bf16_t kc[32 * M];       // row-major 32x128 (B of Qc*Kc^T)
  __shared__ bf16_t kcT[M * KP];      // kcT[m][i]   (A of state update)
  __shared__ bf16_t vcT[DH * VP];     // vcT[d][i]   (B operand, k-contiguous)
  __shared__ bf16_t skvbT[DH * MP];   // skvbT[d][m] (B of inter, k-contiguous)
  __shared__ bf16_t ab[32 * 32];      // masked scores bf16 (A of intra)
  __shared__ float  numb[32 * DH];    // inter numerator
  __shared__ float  skl[M];           // running colsum of K features
  __shared__ float  den[32];

  v8f accS[4];
#pragma unroll
  for (int i = 0; i < 4; ++i) accS[i] = vzero8();
  for (int i = t; i < M; i += 256) skl[i] = 0.f;
  __syncthreads();

  for (int c = 0; c < S / 32; ++c) {
    const int s0 = c * 32;

    // ---- stage chunk (f32 -> bf16, 16B global reads) ----
    for (int i = t; i < 32 * M / 4; i += 256) {  // 1024 chunks of 4
      int r = i >> 5, c4 = (i & 31) * 4;
      size_t g = (((size_t)((b * S + s0 + r) * H + h)) * M + c4) / 4;
      float4 q4 = ((const float4*)Qf)[g];
      float4 k4 = ((const float4*)Kf)[g];
      v4bf qp, kp;
      qp[0] = (bf16_t)q4.x; qp[1] = (bf16_t)q4.y;
      qp[2] = (bf16_t)q4.z; qp[3] = (bf16_t)q4.w;
      kp[0] = (bf16_t)k4.x; kp[1] = (bf16_t)k4.y;
      kp[2] = (bf16_t)k4.z; kp[3] = (bf16_t)k4.w;
      *(v4bf*)(qc + r * M + c4) = qp;
      *(v4bf*)(kc + r * M + c4) = kp;
#pragma unroll
      for (int j = 0; j < 4; ++j) kcT[(c4 + j) * KP + r] = kp[j];
    }
    for (int i = t; i < 32 * DH / 4; i += 256) {  // 512 chunks
      int r = i >> 4, c4 = (i & 15) * 4;
      size_t g = (((size_t)((b * S + s0 + r) * H + h)) * DH + c4) / 4;
      float4 v4 = ((const float4*)Vf)[g];
      vcT[(c4 + 0) * VP + r] = (bf16_t)v4.x;
      vcT[(c4 + 1) * VP + r] = (bf16_t)v4.y;
      vcT[(c4 + 2) * VP + r] = (bf16_t)v4.z;
      vcT[(c4 + 3) * VP + r] = (bf16_t)v4.w;
    }
    // dump Skv accumulators -> skvbT (one 16B store per tile per lane)
    {
      const int n = lane & 15, mo = (lane >> 4) * 8;
#pragma unroll
      for (int ct = 0; ct < 4; ++ct) {
        v8bf pk;
#pragma unroll
        for (int r = 0; r < 8; ++r) pk[r] = (bf16_t)accS[ct][r];
        *(v8bf*)(skvbT + (ct * 16 + n) * MP + wave * 16 + mo) = pk;
      }
    }
    __syncthreads();

    // ---- waves 0-3: A = mask(Qc Kc^T); waves 4-7: inter = Qc * Skv ----
    if (wave < 4) {
      const int ti = wave >> 1, tj = wave & 1;
      v8f acc = vzero8();
#pragma unroll
      for (int k0 = 0; k0 < M; k0 += 32) {
        v16bf a = ld_a_rowmajor(qc + (ti * 16) * M + k0, M);
        v16bf bm = ld_b_kcont(kc + (tj * 16) * M + k0, M);  // Kc^T, k-contig
        acc = wmma_bf16(a, bm, acc);
      }
      const int n = lane & 15, mo = (lane >> 4) * 8;
#pragma unroll
      for (int r = 0; r < 8; ++r) {
        int i = ti * 16 + mo + r, j = tj * 16 + n;
        ab[i * 32 + j] = (bf16_t)((j <= i) ? acc[r] : 0.f);  // causal mask
      }
    } else {
      const int w = wave - 4;
      const int ti = w >> 1;
      const int nc0 = (w & 1) * 32;
#pragma unroll
      for (int sub = 0; sub < 2; ++sub) {
        const int n0 = nc0 + sub * 16;
        v8f acc = vzero8();
#pragma unroll
        for (int k0 = 0; k0 < M; k0 += 32) {
          v16bf a = ld_a_rowmajor(qc + (ti * 16) * M + k0, M);
          v16bf bm = ld_b_kcont(skvbT + n0 * MP + k0, MP);
          acc = wmma_bf16(a, bm, acc);
        }
        const int n = lane & 15, mo = (lane >> 4) * 8;
#pragma unroll
        for (int r = 0; r < 8; ++r)
          numb[(ti * 16 + mo + r) * DH + n0 + n] = acc[r];
      }
    }
    __syncthreads();

    // ---- denominators: den = Qc.sk + rowsum(masked A) ----
    if (t < 32) {
      float dintra = 0.f;
#pragma unroll 8
      for (int j = 0; j < 32; ++j) dintra += (float)ab[t * 32 + j];
      float dinter = 0.f;
#pragma unroll 8
      for (int m = 0; m < M; ++m) dinter += (float)qc[t * M + m] * skl[m];
      den[t] = dinter + dintra + 1e-6f;
    }
    __syncthreads();

    // ---- intra = A * Vc ; out = (inter + intra) / den ----
    {
      const int ti = wave >> 2, n0 = (wave & 3) * 16;
      v8f acc = vzero8();
      v16bf a = ld_a_rowmajor(ab + (ti * 16) * 32, 32);
      v16bf bm = ld_b_kcont(vcT + n0 * VP, VP);
      acc = wmma_bf16(a, bm, acc);
      const int n = lane & 15, mo = (lane >> 4) * 8;
#pragma unroll
      for (int r = 0; r < 8; ++r) {
        int i = ti * 16 + mo + r, d = n0 + n;
        float o = (numb[i * DH + d] + acc[r]) / den[i];
        Out[((size_t)((b * S + s0 + i) * H + h)) * DH + d] = o;
      }
    }

    // ---- state update: Skv += Kc^T * Vc (wave owns feature-row tile) ----
#pragma unroll
    for (int ct = 0; ct < 4; ++ct) {
      v16bf a = ld_a_rowmajor(kcT + (wave * 16) * KP, KP);
      v16bf bm = ld_b_kcont(vcT + ct * 16 * VP, VP);
      accS[ct] = wmma_bf16(a, bm, accS[ct]);
    }
    __syncthreads();

    // ---- sk += colsum(Kc) (contiguous reads via kcT) ----
    for (int m = t; m < M; m += 256) {
      float sadd = 0.f;
#pragma unroll 8
      for (int i = 0; i < 32; ++i) sadd += (float)kcT[m * KP + i];
      skl[m] += sadd;
    }
    __syncthreads();
  }
}

// ---------------------------------------------------------------------------
// Residual add + LayerNorm over DM=512; emits f32 + bf16. Block per row.
// ---------------------------------------------------------------------------
__global__ __launch_bounds__(256) void add_ln_kernel(const float* __restrict__ X,
                                                     const float* __restrict__ Yd,
                                                     const float* __restrict__ g,
                                                     const float* __restrict__ bta,
                                                     float* __restrict__ Of,
                                                     bf16_t* __restrict__ Ob) {
  using namespace cfg;
  const int row = blockIdx.x;
  const int t = threadIdx.x;
  __shared__ float red[256];
  const float* xr = X + (size_t)row * DM;
  const float* yr = Yd + (size_t)row * DM;

  float v0 = xr[t] + yr[t];
  float v1 = xr[t + 256] + yr[t + 256];
  red[t] = v0 + v1;
  __syncthreads();
  for (int s = 128; s > 0; s >>= 1) {
    if (t < s) red[t] += red[t + s];
    __syncthreads();
  }
  const float mu = red[0] * (1.f / DM);
  __syncthreads();
  float d0 = v0 - mu, d1 = v1 - mu;
  red[t] = d0 * d0 + d1 * d1;
  __syncthreads();
  for (int s = 128; s > 0; s >>= 1) {
    if (t < s) red[t] += red[t + s];
    __syncthreads();
  }
  const float rstd = rsqrtf(red[0] * (1.f / DM) + 1e-5f);

  float o0 = d0 * rstd * g[t] + bta[t];
  float o1 = d1 * rstd * g[t + 256] + bta[t + 256];
  Of[(size_t)row * DM + t] = o0;
  Of[(size_t)row * DM + t + 256] = o1;
  Ob[(size_t)row * DM + t] = (bf16_t)o0;
  Ob[(size_t)row * DM + t + 256] = (bf16_t)o1;
}

// ---------------------------------------------------------------------------
// launcher
// ---------------------------------------------------------------------------
extern "C" void kernel_launch(void* const* d_in, const int* in_sizes, int n_in,
                              void* d_out, int out_size, void* d_ws, size_t ws_size,
                              hipStream_t stream) {
  using namespace cfg;
  (void)in_sizes; (void)n_in; (void)out_size; (void)ws_size;

  const float* x       = (const float*)d_in[0];
  const float* Wq      = (const float*)d_in[1];
  const float* bq      = (const float*)d_in[2];
  const float* Wk      = (const float*)d_in[3];
  const float* bk      = (const float*)d_in[4];
  const float* Wv      = (const float*)d_in[5];
  const float* bv      = (const float*)d_in[6];
  const float* Wo      = (const float*)d_in[7];
  const float* bo      = (const float*)d_in[8];
  const float* W1      = (const float*)d_in[9];
  const float* b1      = (const float*)d_in[10];
  const float* W2      = (const float*)d_in[11];
  const float* b2      = (const float*)d_in[12];
  const float* ln1_g   = (const float*)d_in[13];
  const float* ln1_b   = (const float*)d_in[14];
  const float* ln2_g   = (const float*)d_in[15];
  const float* ln2_b   = (const float*)d_in[16];
  const float* sfreq   = (const float*)d_in[17];
  const float* soff    = (const float*)d_in[18];
  const float* sgain   = (const float*)d_in[19];
  const float* snoise  = (const float*)d_in[20];
  const float* gate    = (const float*)d_in[21];
  const float* gnoise  = (const float*)d_in[22];
  const float* omega   = (const float*)d_in[23];

  char* ws = (char*)d_ws;
  size_t off = 0;
  auto alloc = [&](size_t bytes) -> void* {
    off = (off + 255) & ~(size_t)255;
    void* p = ws + off;
    off += bytes;
    return p;
  };

  float*  cur  = (float*)alloc((size_t)NT * DM * 4);
  bf16_t* curb = (bf16_t*)alloc((size_t)NT * DM * 2);
  float*  qf   = (float*)alloc((size_t)NT * DM * 4);
  float*  kf   = (float*)alloc((size_t)NT * DM * 4);
  float*  vf   = (float*)alloc((size_t)NT * DM * 4);
  float*  qbar = (float*)alloc((size_t)S * H * DH * R * 4);
  float*  kbar = (float*)alloc((size_t)S * H * DH * R * 4);
  float*  qh   = (float*)alloc((size_t)NT * H * 64 * 4);
  float*  kh   = (float*)alloc((size_t)NT * H * 64 * 4);
  float*  Qfm  = (float*)alloc((size_t)NT * H * M * 4);
  float*  Kfm  = (float*)alloc((size_t)NT * H * M * 4);
  float*  att  = (float*)alloc((size_t)NT * DM * 4);
  bf16_t* attb = (bf16_t*)alloc((size_t)NT * DM * 2);
  float*  proj = (float*)alloc((size_t)NT * DM * 4);
  float*  hbuf = (float*)alloc((size_t)NT * DM * 4);
  bf16_t* hb   = (bf16_t*)alloc((size_t)NT * DM * 2);
  bf16_t* midb = (bf16_t*)alloc((size_t)NT * DF * 2);
  float*  ybuf = (float*)alloc((size_t)NT * DM * 4);
  bf16_t* Wqb  = (bf16_t*)alloc((size_t)L * DM * DM * 2);
  bf16_t* Wkb  = (bf16_t*)alloc((size_t)L * DM * DM * 2);
  bf16_t* Wvb  = (bf16_t*)alloc((size_t)L * DM * DM * 2);
  bf16_t* Wob  = (bf16_t*)alloc((size_t)L * DM * DM * 2);
  bf16_t* W1b  = (bf16_t*)alloc((size_t)L * DM * DF * 2);
  bf16_t* W2b  = (bf16_t*)alloc((size_t)L * DF * DM * 2);

  auto castN = [&](const float* src, bf16_t* dst, int n) {
    cast_f32_bf16_kernel<<<(n + 255) / 256, 256, 0, stream>>>(src, dst, n);
  };

  // one-time conversions
  hipMemcpyAsync(cur, x, (size_t)NT * DM * 4, hipMemcpyDeviceToDevice, stream);
  castN(x, curb, NT * DM);
  castN(Wq, Wqb, L * DM * DM);
  castN(Wk, Wkb, L * DM * DM);
  castN(Wv, Wvb, L * DM * DM);
  castN(Wo, Wob, L * DM * DM);
  castN(W1, W1b, L * DM * DF);
  castN(W2, W2b, L * DF * DM);

  const int ewBlocks = (NT * H * 64) / 256;  // 8192

  for (int l = 0; l < L; ++l) {
    // SPE positional code
    spe_code_kernel<<<H * DH, 256, 0, stream>>>(
        sfreq + (size_t)l * H * DH * NS, soff + (size_t)l * H * DH * NS,
        sgain + (size_t)l * H * DH * NS, snoise + (size_t)l * H * DH * 2 * NS * R,
        qbar, kbar);

    // QKV projections (bf16 WMMA, f32 accumulate)
    dim3 g512(NT / 128, DM / 128);
    gemm_kernel<0, 0><<<g512, 256, 0, stream>>>(curb, Wqb + (size_t)l * DM * DM,
                                                bq + (size_t)l * DM, qf, nullptr,
                                                NT, DM, DM);
    gemm_kernel<0, 0><<<g512, 256, 0, stream>>>(curb, Wkb + (size_t)l * DM * DM,
                                                bk + (size_t)l * DM, kf, nullptr,
                                                NT, DM, DM);
    gemm_kernel<0, 0><<<g512, 256, 0, stream>>>(curb, Wvb + (size_t)l * DM * DM,
                                                bv + (size_t)l * DM, vf, nullptr,
                                                NT, DM, DM);

    // gated SPE filter -> qh/kh
    spe_filter_kernel<<<ewBlocks, 256, 0, stream>>>(
        qf, kf, qbar, kbar, gate + (size_t)l * H * DH,
        gnoise + (size_t)l * H * DH * R, qh, kh);

    // FAVOR+ features
    favor_kernel<<<ewBlocks, 256, 0, stream>>>(qh, omega + (size_t)l * DH * 64, Qfm);
    favor_kernel<<<ewBlocks, 256, 0, stream>>>(kh, omega + (size_t)l * DH * 64, Kfm);

    // chunked causal linear attention (WMMA)
    attn_kernel<<<B * H, 256, 0, stream>>>(Qfm, Kfm, vf, att);

    // output projection
    castN(att, attb, NT * DM);
    gemm_kernel<0, 0><<<g512, 256, 0, stream>>>(attb, Wob + (size_t)l * DM * DM,
                                                bo + (size_t)l * DM, proj, nullptr,
                                                NT, DM, DM);

    // h = LN(cur + proj)
    add_ln_kernel<<<NT, 256, 0, stream>>>(cur, proj, ln1_g + (size_t)l * DM,
                                          ln1_b + (size_t)l * DM, hbuf, hb);

    // FFN
    dim3 gF(NT / 128, DF / 128);
    gemm_kernel<1, 1><<<gF, 256, 0, stream>>>(hb, W1b + (size_t)l * DM * DF,
                                              b1 + (size_t)l * DF, nullptr, midb,
                                              NT, DF, DM);
    gemm_kernel<0, 0><<<g512, 256, 0, stream>>>(midb, W2b + (size_t)l * DF * DM,
                                                b2 + (size_t)l * DM, ybuf, nullptr,
                                                NT, DM, DF);

    // cur = LN(h + y)  (last layer writes d_out directly)
    float* outf = (l == L - 1) ? (float*)d_out : cur;
    add_ln_kernel<<<NT, 256, 0, stream>>>(hbuf, ybuf, ln2_g + (size_t)l * DM,
                                          ln2_b + (size_t)l * DM, outf, curb);
  }
}
